// Attention_UP_decoder_11373073400043
// MI455X (gfx1250) — matrix operand order repaired and checked by reading, not compiled, not run on hardware
//
#include <hip/hip_runtime.h>

typedef __attribute__((ext_vector_type(16))) _Float16 v16h;
typedef __attribute__((ext_vector_type(8)))  _Float16 v8h;
typedef __attribute__((ext_vector_type(8)))  float    v8f;
typedef int v4i __attribute__((vector_size(16)));   // matches builtin param type

#define TILE_M 128
#define TILE_N 128
#define TILE_K 32
#define LDSS   40   // padded LDS row stride in halfs (80B, 16B aligned)

#define AS1 __attribute__((address_space(1)))
#define AS3 __attribute__((address_space(3)))

#if __has_builtin(__builtin_amdgcn_global_load_async_to_lds_b128) && \
    __has_builtin(__builtin_amdgcn_s_wait_asynccnt)
#define USE_ASYNC 1
#else
#define USE_ASYNC 0
#endif

// ---------------------------------------------------------------------------
// Stage one 128x32 f16 tile of A and one of B into LDS. Each of the 256
// threads moves two aligned 16B chunks of A and two of B (rows of 64B).
// With USE_ASYNC these are 4 async-DMA issues per thread (ASYNCcnt += 4/wave).
// ---------------------------------------------------------------------------
__device__ __forceinline__ void stage_tile(
    const _Float16* __restrict__ Ag, int lda,
    const _Float16* __restrict__ Bg, int ldb,
    int m0, int n0, int k0, int tid,
    _Float16* la_base, _Float16* lb_base)
{
    #pragma unroll
    for (int e = 0; e < 2; ++e) {
        const int idx = tid + e * 256;
        const int rr = idx >> 2;          // 0..127
        const int cc = (idx & 3) * 8;     // 0,8,16,24 halfs
        const _Float16* ga = Ag + (size_t)(m0 + rr) * lda + (k0 + cc);
        const _Float16* gb = Bg + (size_t)(n0 + rr) * ldb + (k0 + cc);
        _Float16* la = la_base + rr * LDSS + cc;
        _Float16* lb = lb_base + rr * LDSS + cc;
#if USE_ASYNC
        __builtin_amdgcn_global_load_async_to_lds_b128(
            (AS1 v4i*)(void*)ga, (AS3 v4i*)(void*)la, 0, 0);
        __builtin_amdgcn_global_load_async_to_lds_b128(
            (AS1 v4i*)(void*)gb, (AS3 v4i*)(void*)lb, 0, 0);
#else
        *(v8h*)la = *(const v8h*)ga;
        *(v8h*)lb = *(const v8h*)gb;
#endif
    }
}

// ---------------------------------------------------------------------------
// Batched GEMM, f16 operands, f32 WMMA accumulate. 128x128 block, 8 waves,
// each wave owns a 32x64 output patch: 2 A-frags x 4 B-frags = 8 WMMAs/K-tile.
//   A[b]: (M x K) f16 row-major, lda (batch stride sA, 0 for shared weights)
//   B[b]: stored TRANSPOSED as (N x K) f16 row-major, ldb
//   C[b]: cmode 0 -> f32 natural [M,N] (optional accum += )
//         cmode 1 -> f16 natural [M,N]
//         cmode 2 -> f16 transposed [N,M]  (ldc = M)
// Epilogues: bias[M] (batch stride sBias), bn(gamma,beta)+relu.
// Requires M%128==0, N%128==0, K%32==0 (holds for every GEMM here).
// Double-buffered LDS; tile k+1 staged via async DMA while tile k feeds WMMAs.
// ---------------------------------------------------------------------------
__global__ __launch_bounds__(256) void gemm_wmma_f16(
    const _Float16* __restrict__ A, long long sA, int lda,
    const _Float16* __restrict__ B, long long sB, int ldb,
    void* __restrict__ Cv, long long sC, int ldc, int cmode, int accum,
    int M, int N, int K,
    const float* __restrict__ bias, long long sBias,
    const float* __restrict__ gamma, const float* __restrict__ beta)
{
    __shared__ _Float16 ldsA[2][TILE_M * LDSS];
    __shared__ _Float16 ldsB[2][TILE_N * LDSS];

    const int bz = blockIdx.z;
    A += (long long)bz * sA;
    B += (long long)bz * sB;
    float*    C32 = (float*)Cv    + (long long)bz * sC;
    _Float16* C16 = (_Float16*)Cv + (long long)bz * sC;
    if (bias) bias += (long long)bz * sBias;

    const int m0 = blockIdx.y * TILE_M;
    const int n0 = blockIdx.x * TILE_N;

    const int tid  = threadIdx.x;     // 256 threads = 8 waves (wave32)
    const int lane = tid & 31;
    const int wave = tid >> 5;
    const int wm   = wave & 3;        // 4 M sub-blocks of 32 rows
    const int wn   = wave >> 2;       // 2 N sub-blocks of 64 cols
    const int r    = lane & 15;
    const int g    = lane >> 4;

    v8f acc[2][4] = {};

    const int nk = K / TILE_K;
    stage_tile(A, lda, B, ldb, m0, n0, 0, tid, ldsA[0], ldsB[0]);

    for (int kt = 0; kt < nk; ++kt) {
        const int cur = kt & 1;
        if (kt + 1 < nk)
            stage_tile(A, lda, B, ldb, m0, n0, (kt + 1) * TILE_K, tid,
                       ldsA[cur ^ 1], ldsB[cur ^ 1]);
#if USE_ASYNC
        if (kt + 1 < nk) __builtin_amdgcn_s_wait_asynccnt(4); // tile kt landed
        else             __builtin_amdgcn_s_wait_asynccnt(0);
#endif
        __syncthreads();

        // A fragments: row = wm*32 + m*16 + r ; elem i -> K = g*8 + (i<8?i:i+8)
        v16h af[2];
        #pragma unroll
        for (int m = 0; m < 2; ++m) {
            const _Float16* pa =
                &ldsA[cur][(wm * 32 + m * 16 + r) * LDSS + g * 8];
            v8h a0 = *(const v8h*)pa;
            v8h a1 = *(const v8h*)(pa + 16);
            #pragma unroll
            for (int i = 0; i < 8; ++i) { af[m][i] = a0[i]; af[m][i + 8] = a1[i]; }
        }
        // B fragments: col = wn*64 + n*16 + r ; elem i -> K = g*8 + (i<8?i:i+8)
        v16h bf[4];
        #pragma unroll
        for (int n = 0; n < 4; ++n) {
            const _Float16* pb =
                &ldsB[cur][(wn * 64 + n * 16 + r) * LDSS + g * 8];
            v8h b0 = *(const v8h*)pb;
            v8h b1 = *(const v8h*)(pb + 16);
            #pragma unroll
            for (int i = 0; i < 8; ++i) { bf[n][i] = b0[i]; bf[n][i + 8] = b1[i]; }
        }
        #pragma unroll
        for (int m = 0; m < 2; ++m)
            #pragma unroll
            for (int n = 0; n < 4; ++n)
                acc[m][n] = __builtin_amdgcn_wmma_f32_16x16x32_f16(
                    false, af[m], false, bf[n], (short)0, acc[m][n],
                    false, false);

        __syncthreads();   // safe to overwrite buffer (cur) next iteration
    }

    // ---- epilogue: C/D layout: elem j -> M = j + 8*g, N = r ----
    const float RSQ = 0.99999500003749971f;   // 1/sqrt(1 + 1e-5)
    #pragma unroll
    for (int m = 0; m < 2; ++m) {
        #pragma unroll
        for (int n = 0; n < 4; ++n) {
            const int col = n0 + wn * 64 + n * 16 + r;
            #pragma unroll
            for (int j = 0; j < 8; ++j) {
                const int row = m0 + wm * 32 + m * 16 + g * 8 + j;
                float v = acc[m][n][j];
                if (gamma) {
                    v = v * (gamma[row] * RSQ) + beta[row];
                    v = v > 0.0f ? v : 0.0f;
                }
                if (bias) v += bias[row];
                if (cmode == 0) {
                    size_t off = (size_t)row * ldc + col;
                    if (accum) v += C32[off];
                    C32[off] = v;
                } else if (cmode == 1) {
                    C16[(size_t)row * ldc + col] = (_Float16)v;
                } else {
                    C16[(size_t)col * ldc + row] = (_Float16)v;
                }
            }
        }
    }
}

// ---------------------------------------------------------------------------
// Elementwise / glue kernels
// ---------------------------------------------------------------------------
__global__ void cvt_f16_k(const float* __restrict__ src,
                          _Float16* __restrict__ dst, long long n)
{
    long long i = (long long)blockIdx.x * blockDim.x + threadIdx.x;
    if (i < n) dst[i] = (_Float16)src[i];
}

// f32 (B,Cc,S) -> f16 transposed (B,S,Cc)
__global__ void transpose_cvt_k(const float* __restrict__ src,
                                _Float16* __restrict__ dst,
                                int Bn, int Cc, int S)
{
    long long i = (long long)blockIdx.x * blockDim.x + threadIdx.x;
    long long total = (long long)Bn * Cc * S;
    if (i >= total) return;
    int s = (int)(i % S); long long t = i / S;
    int c = (int)(t % Cc); int b = (int)(t / Cc);
    dst[((size_t)b * S + s) * Cc + c] = (_Float16)src[i];
}

// bilinear (align_corners=True): f16 natural (B,Cc,IH,IW) ->
// f16 transposed concat slice (B, OH*OW, CT) at channel chOff
__global__ void resize_bilinear_k(const _Float16* __restrict__ src,
                                  _Float16* __restrict__ dst,
                                  int Bn, int Cc, int IH, int IW,
                                  int OH, int OW, int CT, int chOff)
{
    long long i = (long long)blockIdx.x * blockDim.x + threadIdx.x;
    long long total = (long long)Bn * Cc * OH * OW;
    if (i >= total) return;
    int ox = (int)(i % OW); long long t = i / OW;
    int oy = (int)(t % OH); t /= OH;
    int c  = (int)(t % Cc); int b = (int)(t / Cc);

    float ry = (OH > 1) ? (float)(IH - 1) / (float)(OH - 1) : 0.0f;
    float rx = (OW > 1) ? (float)(IW - 1) / (float)(OW - 1) : 0.0f;
    float fy = oy * ry, fx = ox * rx;
    int y0 = (int)fy, x0 = (int)fx;
    int y1 = y0 + 1; if (y1 > IH - 1) y1 = IH - 1;
    int x1 = x0 + 1; if (x1 > IW - 1) x1 = IW - 1;
    float wy = fy - (float)y0, wx = fx - (float)x0;

    const _Float16* s = src + (size_t)(b * Cc + c) * IH * IW;
    float v00 = (float)s[y0 * IW + x0], v01 = (float)s[y0 * IW + x1];
    float v10 = (float)s[y1 * IW + x0], v11 = (float)s[y1 * IW + x1];
    float top = v00 * (1.0f - wx) + v01 * wx;
    float bot = v10 * (1.0f - wx) + v11 * wx;
    float o   = top * (1.0f - wy) + bot * wy;
    dst[((size_t)b * OH * OW + (size_t)(oy * OW + ox)) * CT + chOff + c] =
        (_Float16)o;
}

// f16 natural (B,Cs,S) -> f16 transposed concat slice (B,S,CT) at chOff
__global__ void copy_concat_t_k(const _Float16* __restrict__ src,
                                _Float16* __restrict__ dst,
                                int Bn, int Cs, int S, int CT, int chOff)
{
    long long i = (long long)blockIdx.x * blockDim.x + threadIdx.x;
    long long total = (long long)Bn * Cs * S;
    if (i >= total) return;
    int s = (int)(i % S); long long t = i / S;
    int c = (int)(t % Cs); int b = (int)(t / Cs);
    dst[((size_t)b * S + s) * CT + chOff + c] = src[i];
}

// row softmax: f32 in -> f16 out (one block per row)
__global__ __launch_bounds__(256) void softmax_rows_k(
    const float* __restrict__ in, _Float16* __restrict__ out, int L)
{
    __shared__ float red[256];
    const float* p = in  + (size_t)blockIdx.x * L;
    _Float16*    q = out + (size_t)blockIdx.x * L;
    int tid = threadIdx.x;

    float m = -3.4e38f;
    for (int i = tid; i < L; i += 256) m = fmaxf(m, p[i]);
    red[tid] = m; __syncthreads();
    for (int s = 128; s > 0; s >>= 1) {
        if (tid < s) red[tid] = fmaxf(red[tid], red[tid + s]);
        __syncthreads();
    }
    m = red[0]; __syncthreads();

    float sum = 0.0f;
    for (int i = tid; i < L; i += 256) sum += __expf(p[i] - m);
    red[tid] = sum; __syncthreads();
    for (int s = 128; s > 0; s >>= 1) {
        if (tid < s) red[tid] += red[tid + s];
        __syncthreads();
    }
    float inv = 1.0f / red[0];
    for (int i = tid; i < L; i += 256)
        q[i] = (_Float16)(__expf(p[i] - m) * inv);
}

// row mean of f16 matrix -> f32 (one block per row)
__global__ __launch_bounds__(256) void mean_rows_k(
    const _Float16* __restrict__ buf, float* __restrict__ out, int L)
{
    __shared__ float red[256];
    const _Float16* p = buf + (size_t)blockIdx.x * L;
    int tid = threadIdx.x;
    float s = 0.0f;
    for (int i = tid; i < L; i += 256) s += (float)p[i];
    red[tid] = s; __syncthreads();
    for (int t = 128; t > 0; t >>= 1) {
        if (tid < t) red[tid] += red[tid + t];
        __syncthreads();
    }
    if (tid == 0) out[blockIdx.x] = red[0] / (float)L;
}

// bias2[b,m] = b_W[m] + sum_k w_W[m,k] * pool[b,k]   (one block per (b,m))
__global__ __launch_bounds__(256) void bias2_k(
    const float* __restrict__ wW, const float* __restrict__ pool,
    const float* __restrict__ bW, float* __restrict__ bias2,
    int NC, int K)
{
    __shared__ float red[256];
    int b = blockIdx.x / NC;
    int m = blockIdx.x % NC;
    int tid = threadIdx.x;
    const float* wr = wW + (size_t)m * K;
    const float* pr = pool + (size_t)b * K;
    float s = 0.0f;
    for (int k = tid; k < K; k += 256) s += wr[k] * pr[k];
    red[tid] = s; __syncthreads();
    for (int t = 128; t > 0; t >>= 1) {
        if (tid < t) red[tid] += red[tid + t];
        __syncthreads();
    }
    if (tid == 0) bias2[blockIdx.x] = red[0] + bW[m];
}

// ---------------------------------------------------------------------------
extern "C" void kernel_launch(void* const* d_in, const int* in_sizes, int n_in,
                              void* d_out, int out_size, void* d_ws, size_t ws_size,
                              hipStream_t stream) {
    const float* x_h    = (const float*)d_in[0];
    const float* x_l    = (const float*)d_in[1];
    const float* w_in_h = (const float*)d_in[2];
    const float* g_in_h = (const float*)d_in[3];
    const float* b_in_h = (const float*)d_in[4];
    const float* w_in_l = (const float*)d_in[5];
    const float* g_in_l = (const float*)d_in[6];
    const float* b_in_l = (const float*)d_in[7];
    const float* w_B    = (const float*)d_in[8];
    const float* b_B    = (const float*)d_in[9];
    const float* w_A    = (const float*)d_in[10];
    const float* b_A    = (const float*)d_in[11];
    const float* w_G    = (const float*)d_in[12];
    const float* b_G    = (const float*)d_in[13];
    const float* w_W    = (const float*)d_in[14];
    const float* b_W    = (const float*)d_in[15];
    const float* w_out  = (const float*)d_in[16];
    const float* b_out  = (const float*)d_in[17];
    float* out = (float*)d_out;

    const int Bn = 4, CH = 512, CL = 256, Cc = 768, NC = 512;
    const int Hh = 48, Wh = 48, Hl = 96, Wl = 96;
    const int Sh = Hh * Wh;   // 2304
    const int Sl = Hl * Wl;   // 9216

    char* wsb = (char*)d_ws;
    size_t off = 0;
    auto allocH = [&](size_t n) {
        off = (off + 255) & ~(size_t)255;
        _Float16* p = (_Float16*)(wsb + off); off += n * sizeof(_Float16);
        return p;
    };
    auto allocF = [&](size_t n) {
        off = (off + 255) & ~(size_t)255;
        float* p = (float*)(wsb + off); off += n * sizeof(float);
        return p;
    };

    // f16 weights
    _Float16* wh16   = allocH((size_t)CH * CH);
    _Float16* wl16   = allocH((size_t)CL * CL);
    _Float16* wA16   = allocH((size_t)NC * Cc);
    _Float16* wB16   = allocH((size_t)Cc * Cc);
    _Float16* wG16   = allocH((size_t)Cc * Cc);
    _Float16* wW16   = allocH((size_t)NC * Cc);
    _Float16* wout16 = allocH((size_t)CL * 2 * Cc);
    // f16 activations (B operands stored transposed [spatial, channel])
    _Float16* xh_t   = allocH((size_t)Bn * Sh * CH);
    _Float16* xl_t   = allocH((size_t)Bn * Sl * CL);
    _Float16* xh1    = allocH((size_t)Bn * CH * Sh);  // natural
    _Float16* xl1    = allocH((size_t)Bn * CL * Sl);  // natural
    _Float16* mdeep_t= allocH((size_t)Bn * Sh * Cc);
    _Float16* mlow_t = allocH((size_t)Bn * Sl * Cc);
    _Float16* Asm16  = allocH((size_t)Bn * NC * Sh);  // natural == [N,K] operand
    _Float16* Bm16   = allocH((size_t)Bn * Cc * Sh);  // natural (A operand)
    _Float16* cw16   = allocH((size_t)Bn * Cc * NC);  // natural (A operand)
    _Float16* Gt16   = allocH((size_t)Bn * Sl * Cc);  // transposed
    _Float16* Wtt16  = allocH((size_t)Bn * Sl * NC);  // transposed
    _Float16* ft16   = allocH((size_t)Bn * Sl * Cc);  // transposed
    // f32 scratch
    float* Al    = allocF((size_t)Bn * NC * Sh);
    float* pool  = allocF((size_t)Bn * Cc);
    float* bias2 = allocF((size_t)Bn * NC);

    dim3 blk(256);
    auto cdiv = [](long long a, long long b) { return (unsigned)((a + b - 1) / b); };
    auto cvt = [&](const float* s, _Float16* d, long long n) {
        cvt_f16_k<<<cdiv(n, 256), blk, 0, stream>>>(s, d, n);
    };
    auto gemm = [&](const _Float16* Ap, long long sA, int lda,
                    const _Float16* Bp, long long sB, int ldb,
                    void* Cp, long long sC, int ldc, int cmode, int accum,
                    int M, int N, int K,
                    const float* bias, long long sBias,
                    const float* gamma, const float* beta) {
        dim3 grid((unsigned)(N / TILE_N), (unsigned)(M / TILE_M), (unsigned)Bn);
        gemm_wmma_f16<<<grid, blk, 0, stream>>>(Ap, sA, lda, Bp, sB, ldb,
                                                Cp, sC, ldc, cmode, accum,
                                                M, N, K, bias, sBias, gamma, beta);
    };

    // ---- 0) one-time f16 conversions ----
    cvt(w_in_h, wh16,   (long long)CH * CH);
    cvt(w_in_l, wl16,   (long long)CL * CL);
    cvt(w_A,    wA16,   (long long)NC * Cc);
    cvt(w_B,    wB16,   (long long)Cc * Cc);
    cvt(w_G,    wG16,   (long long)Cc * Cc);
    cvt(w_W,    wW16,   (long long)NC * Cc);
    cvt(w_out,  wout16, (long long)CL * 2 * Cc);
    transpose_cvt_k<<<cdiv((long long)Bn * CH * Sh, 256), blk, 0, stream>>>(
        x_h, xh_t, Bn, CH, Sh);
    transpose_cvt_k<<<cdiv((long long)Bn * CL * Sl, 256), blk, 0, stream>>>(
        x_l, xl_t, Bn, CL, Sl);

    // ---- 1) x_h' = relu(bn(w_in_h @ x_h))   M=512 N=2304 K=512, f16 natural
    gemm(wh16, 0, CH, xh_t, (long long)Sh * CH, CH,
         xh1, (long long)CH * Sh, Sh, 1, 0, CH, Sh, CH,
         nullptr, 0, g_in_h, b_in_h);
    // ---- 2) x_l' = relu(bn(w_in_l @ x_l))   M=256 N=9216 K=256
    gemm(wl16, 0, CL, xl_t, (long long)Sl * CL, CL,
         xl1, (long long)CL * Sl, Sl, 1, 0, CL, Sl, CL,
         nullptr, 0, g_in_l, b_in_l);

    // ---- 3..6) build m_deep^T (B,Sh,768) and m_low^T (B,Sl,768) ----
    resize_bilinear_k<<<cdiv((long long)Bn * CH * Sl, 256), blk, 0, stream>>>(
        xh1, mlow_t, Bn, CH, Hh, Wh, Hl, Wl, Cc, CL);     // upsample x_h'
    resize_bilinear_k<<<cdiv((long long)Bn * CL * Sh, 256), blk, 0, stream>>>(
        xl1, mdeep_t, Bn, CL, Hl, Wl, Hh, Wh, Cc, 0);     // downsample x_l'
    copy_concat_t_k<<<cdiv((long long)Bn * CH * Sh, 256), blk, 0, stream>>>(
        xh1, mdeep_t, Bn, CH, Sh, Cc, CL);
    copy_concat_t_k<<<cdiv((long long)Bn * CL * Sl, 256), blk, 0, stream>>>(
        xl1, mlow_t, Bn, CL, Sl, Cc, 0);

    // ---- 7) A logits (f32) = w_A @ m_deep + b_A   M=512 N=2304 K=768
    gemm(wA16, 0, Cc, mdeep_t, (long long)Sh * Cc, Cc,
         Al, (long long)NC * Sh, Sh, 0, 0, NC, Sh, Cc,
         b_A, 0, nullptr, nullptr);
    // ---- 8) softmax over spatial -> f16 [512,2304] (the [N,K] operand)
    softmax_rows_k<<<Bn * NC, blk, 0, stream>>>(Al, Asm16, Sh);

    // ---- 9) B = w_B @ m_deep + b_B   M=768 N=2304 K=768, f16 natural
    gemm(wB16, 0, Cc, mdeep_t, (long long)Sh * Cc, Cc,
         Bm16, (long long)Cc * Sh, Sh, 1, 0, Cc, Sh, Cc,
         b_B, 0, nullptr, nullptr);
    // ---- 10) pool + fused bias2 = w_W @ pool + b_W
    mean_rows_k<<<Bn * Cc, blk, 0, stream>>>(Bm16, pool, Sh);
    bias2_k<<<Bn * NC, blk, 0, stream>>>(w_W, pool, b_W, bias2, NC, Cc);

    // ---- 11) code_word = B_ @ A^T (batched)  M=768 N=512 K=2304, f16 natural
    gemm(Bm16, (long long)Cc * Sh, Sh, Asm16, (long long)NC * Sh, Sh,
         cw16, (long long)Cc * NC, NC, 1, 0, Cc, NC, Sh,
         nullptr, 0, nullptr, nullptr);

    // ---- 12) G = w_G @ m_low + b_G   M=768 N=9216 K=768, f16 transposed
    gemm(wG16, 0, Cc, mlow_t, (long long)Sl * Cc, Cc,
         Gt16, (long long)Sl * Cc, Cc, 2, 0, Cc, Sl, Cc,
         b_G, 0, nullptr, nullptr);

    // ---- 13) Wt = w_W @ G + bias2[b]   M=512 N=9216 K=768, f16 transposed
    gemm(wW16, 0, Cc, Gt16, (long long)Sl * Cc, Cc,
         Wtt16, (long long)Sl * NC, NC, 2, 0, NC, Sl, Cc,
         bias2, NC, nullptr, nullptr);

    // ---- 14) f = code_word @ Wt (batched)  M=768 N=9216 K=512, f16 transposed
    gemm(cw16, (long long)Cc * NC, NC, Wtt16, (long long)Sl * NC, NC,
         ft16, (long long)Sl * Cc, Cc, 2, 0, Cc, Sl, NC,
         nullptr, 0, nullptr, nullptr);

    // ---- 15) out = w_out[:, :768] @ f + b_out   M=256 N=9216 K=768, f32
    gemm(wout16, 0, 2 * Cc, ft16, (long long)Sl * Cc, Cc,
         out, (long long)CL * Sl, Sl, 0, 0, CL, Sl, Cc,
         b_out, 0, nullptr, nullptr);
    // ---- 16) out += w_out[:, 768:] @ G   (accumulate)
    gemm(wout16 + Cc, 0, 2 * Cc, Gt16, (long long)Sl * Cc, Cc,
         out, (long long)CL * Sl, Sl, 0, 1, CL, Sl, Cc,
         nullptr, 0, nullptr, nullptr);
}
